// ELSAAttention_58506044506324
// MI455X (gfx1250) — matrix operand order repaired and checked by reading, not compiled, not run on hardware
//
#include <hip/hip_runtime.h>
#include <hip/hip_bf16.h>
#include <math.h>

// ---------------- CDNA5 WMMA types ----------------
typedef __attribute__((ext_vector_type(16))) _Float16 v16h;
typedef __attribute__((ext_vector_type(8)))  float    v8f;
typedef __attribute__((ext_vector_type(4)))  unsigned u32x4;

struct alignas(32) Frag32B { u32x4 a, b; };

__device__ __forceinline__ v16h load_frag16(const _Float16* p0, const _Float16* p1) {
  Frag32B f;
  f.a = *reinterpret_cast<const u32x4*>(p0);
  f.b = *reinterpret_cast<const u32x4*>(p1);
  return __builtin_bit_cast(v16h, f);
}

// ---------------- problem constants ----------------
#define BB      32
#define NN      196
#define HH      14
#define DIM     384
#define DIMQK   256
#define DIMV    384
#define QKVO    704      // (multiple of 64)
#define HEADS   12
#define CG      32       // DIMV / HEADS
#define KS      7
#define TAPS    49
#define PWOUT   588      // 49*12
#define PWPAD   640      // padded to multiple of 64
#define MTOT    (BB*NN)  // 6272 = 392*16
#define SCALE_F 0.17677669529663687f  // 32^-0.5

// =====================================================================
// WMMA GEMM:  C[M][Nld] = A[M][K](f16,row-major,K-contig) *
//             W[Nld][K]^T (f16, (out,in) row-major) + bias
// One wave (32 threads) per block computes a 16 x (16*NT) tile.
// =====================================================================
template<int K, int NT>
__global__ void __launch_bounds__(32)
gemm_wmma_f16(const _Float16* __restrict__ A, const _Float16* __restrict__ W,
              const float* __restrict__ bias, float* __restrict__ C,
              int Nld, int Nreal) {
  const int m0   = blockIdx.x << 4;
  const int n0   = blockIdx.y * (NT << 4);
  const int lane = threadIdx.x;
  const int col  = lane & 15;   // A-row / B-col / C-col within the 16-tile
  const int hi   = lane >> 4;   // selects which K-half this lane holds

  // A fragment (16x32 f16): lanes 0-15 hold K{0..7,16..23}, lanes 16-31 K{8..15,24..31}
  const _Float16* pa = A + (long)(m0 + col) * K + hi * 8;

  v8f acc[NT] = {};

  for (int k0 = 0; k0 < K; k0 += 32) {
    v16h av = load_frag16(pa + k0, pa + k0 + 16);
    __builtin_prefetch(pa + k0 + 64, 0, 0);   // global_prefetch_b8 next A tile
#pragma unroll
    for (int t = 0; t < NT; ++t) {
      // B fragment (32x16): lanes 0-15 hold K{0..7,16..23} of column `col`, lanes 16-31 K{8..15,24..31}
      const _Float16* pb = W + (long)(n0 + (t << 4) + col) * K + k0 + hi * 8;
      v16h bv = load_frag16(pb, pb + 16);
      acc[t] = __builtin_amdgcn_wmma_f32_16x16x32_f16(
          false, av, false, bv, (short)0, acc[t], false, false);
    }
  }

#pragma unroll
  for (int t = 0; t < NT; ++t) {
    const int n  = n0 + (t << 4) + col;
    const float bv = (n < Nreal) ? bias[n] : 0.f;
#pragma unroll
    for (int i = 0; i < 8; ++i) {           // C/D: VGPR i -> row 8*hi + i
      const int r = m0 + hi * 8 + i;
      C[(long)r * Nld + n] = acc[t][i] + bv;
    }
  }
}

// ---------------- elementwise prep ----------------
__global__ void cast_f16_kernel(const float* __restrict__ s, _Float16* __restrict__ d, int n) {
  int i = blockIdx.x * 256 + threadIdx.x;
  if (i < n) d[i] = (_Float16)s[i];
}

// pad w_pw (588x256) -> (640x256) f16, zero pad rows
__global__ void cast_pw_kernel(const float* __restrict__ s, _Float16* __restrict__ d) {
  int i = blockIdx.x * 256 + threadIdx.x;
  if (i >= PWPAD * DIMQK) return;
  int o = i >> 8, c = i & 255;
  d[i] = (o < PWOUT) ? (_Float16)s[o * DIMQK + c] : (_Float16)0.f;
}

// hp[b][c][p] = q*k*SCALE, channel-planar for the spatial conv
__global__ void hadamard_kernel(const float* __restrict__ qkv, float* __restrict__ hp) {
  int i = blockIdx.x * 256 + threadIdx.x;
  if (i >= BB * DIMQK * NN) return;
  int b = i / (DIMQK * NN);
  int r = i - b * (DIMQK * NN);
  int c = r / NN, p = r - c * NN;
  const float* row = qkv + ((long)b * NN + p) * QKVO;
  hp[i] = row[c] * row[DIMQK + c] * SCALE_F;
}

// ---------------- grouped 7x7 conv + exact GELU ----------------
// one block per (b, group); writes token-major f16 (K-contig for pw GEMM)
__global__ void __launch_bounds__(256)
conv_dw_gelu_kernel(const float* __restrict__ hp, const float* __restrict__ w_dw,
                    const float* __restrict__ b_dw, _Float16* __restrict__ h_t) {
  const int b = blockIdx.x >> 5;
  const int g = blockIdx.x & 31;
  const int tid = threadIdx.x;
  __shared__ float sIn[8 * 20 * 20];   // zero-padded input tile
  __shared__ float sW[8 * 8 * TAPS];   // 8 oc x 8 ic x 49

  for (int i = tid; i < 8 * 8 * TAPS; i += 256) sW[i] = w_dw[(long)g * 8 * 8 * TAPS + i];
  for (int i = tid; i < 8 * 400; i += 256) sIn[i] = 0.f;
  __syncthreads();
  for (int i = tid; i < 8 * NN; i += 256) {
    int ic = i / NN, p = i - ic * NN, y = p / HH, x = p - y * HH;
    sIn[ic * 400 + (y + 3) * 20 + (x + 3)] = hp[((long)b * DIMQK + g * 8 + ic) * NN + p];
  }
  __syncthreads();

  for (int i = tid; i < 8 * NN; i += 256) {
    int oc = i / NN, p = i - oc * NN, y = p / HH, x = p - y * HH;
    float s = b_dw[g * 8 + oc];
    const float* wrow = &sW[oc * 8 * TAPS];
    for (int ic = 0; ic < 8; ++ic) {
      const float* in = &sIn[ic * 400];
      const float* w  = &wrow[ic * TAPS];
#pragma unroll
      for (int ky = 0; ky < KS; ++ky)
#pragma unroll
        for (int kx = 0; kx < KS; ++kx)
          s = fmaf(in[(y + ky) * 20 + (x + kx)], w[ky * KS + kx], s);
    }
    float gl = 0.5f * s * (1.f + erff(s * 0.70710678118654752f));
    h_t[((long)b * NN + p) * DIMQK + g * 8 + oc] = (_Float16)gl;
  }
}

// ---------------- softmax over the 49 contiguous taps ----------------
__global__ void softmax49_kernel(float* __restrict__ attn) {
  long i = (long)blockIdx.x * 256 + threadIdx.x;
  if (i >= (long)BB * NN * HEADS) return;
  int head = (int)(i % HEADS);
  long bp = i / HEADS;
  float* p = attn + bp * PWPAD + head * TAPS;
  float m = p[0];
#pragma unroll
  for (int j = 1; j < TAPS; ++j) m = fmaxf(m, p[j]);
  float s = 0.f;
#pragma unroll
  for (int j = 0; j < TAPS; ++j) { float e = __expf(p[j] - m); p[j] = e; s += e; }
  float r = 1.f / s;
#pragma unroll
  for (int j = 0; j < TAPS; ++j) p[j] *= r;
}

// ---------------- elsa windowed weighted sum ----------------
// block = (b, head); dynamic LDS: attn slab + padded v tile + ghost mul/add
__global__ void __launch_bounds__(256)
elsa_kernel(const float* __restrict__ qkv, const float* __restrict__ attn,
            const float* __restrict__ ghost, _Float16* __restrict__ out_t) {
  extern __shared__ float sm[];
  float* sAttn = sm;                    // 196*49 = 9604
  float* sV    = sAttn + NN * TAPS;     // 32*400 = 12800 (zero padded 20x20)
  float* sMul  = sV + CG * 400;         // 32*49
  float* sAdd  = sMul + CG * TAPS;      // 32*49
  const int b = blockIdx.x, head = blockIdx.y, tid = threadIdx.x;

  for (int i = tid; i < NN * TAPS; i += 256) {
    int p = i / TAPS, j = i - p * TAPS;
    sAttn[i] = attn[((long)b * NN + p) * PWPAD + head * TAPS + j];
  }
  for (int i = tid; i < CG * 400; i += 256) sV[i] = 0.f;
  for (int i = tid; i < CG * TAPS; i += 256) {
    int cg = i / TAPS, j = i - cg * TAPS;
    int c = head * CG + cg;
    sMul[i] = ghost[(long)c * TAPS + j];                      // LAM == 1
    sAdd[i] = ghost[(long)DIMV * TAPS + c * TAPS + j];        // GAMMA == 1
  }
  __syncthreads();
  for (int i = tid; i < CG * NN; i += 256) {
    int cg = i / NN, p = i - cg * NN, y = p / HH, x = p - y * HH;
    sV[cg * 400 + (y + 3) * 20 + (x + 3)] =
        qkv[((long)b * NN + p) * QKVO + 2 * DIMQK + head * CG + cg];
  }
  __syncthreads();

  for (int i = tid; i < CG * NN; i += 256) {
    int cg = i / NN, p = i - cg * NN, y = p / HH, x = p - y * HH;
    const float* vm = &sV[cg * 400];
    const float* gm = &sMul[cg * TAPS];
    const float* ga = &sAdd[cg * TAPS];
    const float* at = &sAttn[p * TAPS];
    float s = 0.f;
#pragma unroll
    for (int j = 0; j < TAPS; ++j) {
      int dy = j / KS, dx = j - dy * KS;
      s = fmaf(vm[(y + dy) * 20 + (x + dx)], fmaf(gm[j], at[j], ga[j]), s);
    }
    out_t[((long)b * NN + p) * DIMV + head * CG + cg] = (_Float16)s;
  }
}

// =====================================================================
// host launch
// =====================================================================
extern "C" void kernel_launch(void* const* d_in, const int* in_sizes, int n_in,
                              void* d_out, int out_size, void* d_ws, size_t ws_size,
                              hipStream_t stream) {
  const float* x      = (const float*)d_in[0];
  const float* w_qkv  = (const float*)d_in[1];
  const float* b_qkv  = (const float*)d_in[2];
  const float* w_dw   = (const float*)d_in[3];
  const float* b_dw   = (const float*)d_in[4];
  const float* w_pw   = (const float*)d_in[5];
  const float* b_pw   = (const float*)d_in[6];
  const float* ghost  = (const float*)d_in[7];
  const float* w_proj = (const float*)d_in[8];
  const float* b_proj = (const float*)d_in[9];
  float* y = (float*)d_out;

  char* ws = (char*)d_ws;
  size_t off = 0;
  auto alloc = [&](size_t bytes) -> char* {
    char* p = ws + off;
    off = (off + bytes + 255) & ~(size_t)255;
    return p;
  };
  _Float16* x_h     = (_Float16*)alloc((size_t)MTOT * DIM * 2);
  _Float16* wqkv_h  = (_Float16*)alloc((size_t)QKVO * DIM * 2);
  _Float16* wpw_h   = (_Float16*)alloc((size_t)PWPAD * DIMQK * 2);
  _Float16* wproj_h = (_Float16*)alloc((size_t)DIM * DIMV * 2);
  float*    qkv     = (float*)   alloc((size_t)MTOT * QKVO * 4);
  float*    hp      = (float*)   alloc((size_t)BB * DIMQK * NN * 4);
  _Float16* h_t     = (_Float16*)alloc((size_t)MTOT * DIMQK * 2);
  float*    attn    = (float*)   alloc((size_t)MTOT * PWPAD * 4);
  _Float16* out_t   = (_Float16*)alloc((size_t)MTOT * DIMV * 2);
  (void)ws_size; (void)in_sizes; (void)n_in; (void)out_size;

  auto nb = [](long n) { return (int)((n + 255) / 256); };

  // 1) precision prep
  cast_f16_kernel<<<nb((long)MTOT * DIM), 256, 0, stream>>>(x, x_h, MTOT * DIM);
  cast_f16_kernel<<<nb((long)QKVO * DIM), 256, 0, stream>>>(w_qkv, wqkv_h, QKVO * DIM);
  cast_pw_kernel <<<nb((long)PWPAD * DIMQK), 256, 0, stream>>>(w_pw, wpw_h);
  cast_f16_kernel<<<nb((long)DIM * DIMV), 256, 0, stream>>>(w_proj, wproj_h, DIM * DIMV);

  // 2) qkv = x @ w_qkv^T + b_qkv        (WMMA)
  gemm_wmma_f16<DIM, 4><<<dim3(MTOT / 16, QKVO / 64), 32, 0, stream>>>(
      x_h, wqkv_h, b_qkv, qkv, QKVO, QKVO);

  // 3) hp = q*k*scale (channel planar)
  hadamard_kernel<<<nb((long)BB * DIMQK * NN), 256, 0, stream>>>(qkv, hp);

  // 4) grouped 7x7 conv + GELU -> token-major f16
  conv_dw_gelu_kernel<<<BB * (DIMQK / 8), 256, 0, stream>>>(hp, w_dw, b_dw, h_t);

  // 5) attn_raw = h @ w_pw^T + b_pw     (WMMA, N padded 588 -> 640)
  gemm_wmma_f16<DIMQK, 4><<<dim3(MTOT / 16, PWPAD / 64), 32, 0, stream>>>(
      h_t, wpw_h, b_pw, attn, PWPAD, PWOUT);

  // 6) softmax over 49 taps
  softmax49_kernel<<<nb((long)BB * NN * HEADS), 256, 0, stream>>>(attn);

  // 7) elsa windowed weighted sum -> token-major f16
  size_t elsa_lds = (size_t)(NN * TAPS + CG * 400 + 2 * CG * TAPS) * 4;  // ~102 KB
  elsa_kernel<<<dim3(BB, HEADS), 256, elsa_lds, stream>>>(qkv, attn, ghost, out_t);

  // 8) y = out @ w_proj^T + b_proj      (WMMA)
  gemm_wmma_f16<DIMV, 4><<<dim3(MTOT / 16, DIM / 64), 32, 0, stream>>>(
      out_t, wproj_h, b_proj, y, DIM, DIM);
}
